// FraudGNN_61151744361108
// MI455X (gfx1250) — matrix operand order repaired and checked by reading, not compiled, not run on hardware
//
#include <hip/hip_runtime.h>

// GraphSAGE fraud-GNN forward for MI455X (gfx1250), wave32.
// Pipeline: zero -> pack f16 weights -> edge scatter (HW f32 atomics, L2-resident)
//        -> fused [mean|x] WMMA GEMM + bias + ReLU (layer1, K=64)
//        -> edge scatter (64ch) -> fused WMMA GEMM (layer2, K=128) -> output dot.

typedef __attribute__((ext_vector_type(16))) _Float16 v16h;
typedef __attribute__((ext_vector_type(8)))  float    v8f;

#define WMMA(a, b, c) __builtin_amdgcn_wmma_f32_16x16x32_f16(false, (a), false, (b), (short)0, (c), false, false)

__global__ void k_zero(float* __restrict__ p, int n) {
  int i = blockIdx.x * blockDim.x + threadIdx.x;
  int stride = gridDim.x * blockDim.x;
  for (; i < n; i += stride) p[i] = 0.0f;
}

// Pack concatenated f16 weights:
//  wc1[64 rows][64]  : k<30 -> W1l, 32<=k<62 -> W1r, else 0   (layer1 K = [mean(30)+pad | x(30)+pad])
//  wc2[32 rows][128] : k<64 -> W2l, else W2r                  (layer2 K = [mean(64) | h1(64)])
__global__ void k_prep(const float* __restrict__ W1l, const float* __restrict__ W1r,
                       const float* __restrict__ W2l, const float* __restrict__ W2r,
                       _Float16* __restrict__ wc1, _Float16* __restrict__ wc2) {
  for (int t = threadIdx.x; t < 64 * 64; t += blockDim.x) {
    int r = t >> 6, k = t & 63;
    float v = 0.0f;
    if (k < 30) v = W1l[r * 30 + k];
    else if (k >= 32 && k < 62) v = W1r[r * 30 + (k - 32)];
    wc1[t] = (_Float16)v;
  }
  for (int t = threadIdx.x; t < 32 * 128; t += blockDim.x) {
    int r = t >> 7, k = t & 127;
    float v = (k < 64) ? W2l[r * 64 + k] : W2r[r * 64 + (k - 64)];
    wc2[t] = (_Float16)v;
  }
}

// One lane per (edge, channel-slot): c<30 scatters x[src][c], c==31 counts degree.
__global__ void k_scatter1(const float* __restrict__ x, const int* __restrict__ src,
                           const int* __restrict__ dst, float* __restrict__ agg1,
                           float* __restrict__ cnt, int total) {
  int tid = blockIdx.x * blockDim.x + threadIdx.x;
  if (tid >= total) return;
  int e = tid >> 5, c = tid & 31;
  int s = src[e], d = dst[e];
  if (c < 30)       unsafeAtomicAdd(&agg1[d * 32 + c], x[s * 30 + c]);
  else if (c == 31) unsafeAtomicAdd(&cnt[d], 1.0f);
}

__global__ void k_scatter2(const float* __restrict__ h1, const int* __restrict__ src,
                           const int* __restrict__ dst, float* __restrict__ agg2, int total) {
  int tid = blockIdx.x * blockDim.x + threadIdx.x;
  if (tid >= total) return;
  int e = tid >> 6, c = tid & 63;
  unsafeAtomicAdd(&agg2[dst[e] * 64 + c], h1[src[e] * 64 + c]);
}

// Layer 1: h1[16 nodes][64] = relu([mean|x](16x64) @ wc1^T + b1). One wave per 16-node tile,
// 4 N-tiles x 2 K-steps = 8 v_wmma_f32_16x16x32_f16 per wave.
__global__ void k_gemm1(const float* __restrict__ x, const float* __restrict__ agg1,
                        const float* __restrict__ cnt, const _Float16* __restrict__ wc1,
                        const float* __restrict__ bias1, float* __restrict__ h1, int ntiles) {
  const int lane = threadIdx.x & 31;
  const int tile = blockIdx.x * (blockDim.x >> 5) + (threadIdx.x >> 5);
  if (tile >= ntiles) return;                 // wave-uniform: EXEC stays all-ones
  const int base = tile * 16;
  const int row  = lane & 15;                 // A-row / B-col / C-col index
  const int hi   = (lane >> 4) & 1;
  const int aoff = hi ? 16 : 0;               // A K-offset: contiguous k = 16*hi + i (same order as B)
  const int kh   = hi ? 16 : 0;               // B K-half: contiguous k = 16*hi + s (same order as A)
  const int node = base + row;
  const float inv = 1.0f / fmaxf(cnt[node], 1.0f);
  const float* ag = agg1 + node * 32;         // cols 30,31 are zero
  const float* xr = x + (size_t)node * 30;

  v16h a0, a1;                                // K-step 0 (mean), K-step 1 (x)
#pragma unroll
  for (int i = 0; i < 8; ++i) {
    a0[i]     = (_Float16)(ag[aoff + i] * inv);
    a0[i + 8] = (_Float16)(ag[aoff + 8 + i] * inv);
    a1[i]     = (_Float16)xr[aoff + i];
    int kx = aoff + 8 + i;
    int kxc = kx < 30 ? kx : 0;               // clamp: keep load unconditional & in-bounds
    float vx = xr[kxc];
    a1[i + 8] = (_Float16)(kx < 30 ? vx : 0.0f);
  }

  v8f acc[4] = {};
#pragma unroll
  for (int nt = 0; nt < 4; ++nt) {
    const _Float16* wr = wc1 + (nt * 16 + row) * 64;   // row-major weight row == B column
    v16h bA = *(const v16h*)(wr + kh);                 // 32B-aligned
    v16h bB = *(const v16h*)(wr + 32 + kh);
    acc[nt] = WMMA(a0, bA, acc[nt]);
    acc[nt] = WMMA(a1, bB, acc[nt]);
  }

  const int mo = hi ? 8 : 0;
#pragma unroll
  for (int nt = 0; nt < 4; ++nt) {
    const int col = nt * 16 + row;
    const float bv = bias1[col];
#pragma unroll
    for (int j = 0; j < 8; ++j) {
      const int m = base + mo + j;            // C layout: VGPR j -> M = j + 8*hi
      h1[m * 64 + col] = fmaxf(acc[nt][j] + bv, 0.0f);
    }
  }
}

// Layer 2: h2[16][32] = relu([mean2|h1](16x128) @ wc2^T + b2). 2 N-tiles x 4 K-steps = 8 WMMAs.
__global__ void k_gemm2(const float* __restrict__ h1, const float* __restrict__ agg2,
                        const float* __restrict__ cnt, const _Float16* __restrict__ wc2,
                        const float* __restrict__ bias2, float* __restrict__ h2, int ntiles) {
  const int lane = threadIdx.x & 31;
  const int tile = blockIdx.x * (blockDim.x >> 5) + (threadIdx.x >> 5);
  if (tile >= ntiles) return;
  const int base = tile * 16;
  const int row  = lane & 15;
  const int hi   = (lane >> 4) & 1;
  const int aoff = hi ? 16 : 0;
  const int kh   = hi ? 16 : 0;
  const int node = base + row;
  const float inv = 1.0f / fmaxf(cnt[node], 1.0f);
  const float* ag = agg2 + node * 64;
  const float* hr = h1 + node * 64;

  v16h a[4];                                  // steps 0,1: mean2; steps 2,3: h1
#pragma unroll
  for (int s = 0; s < 2; ++s) {
#pragma unroll
    for (int i = 0; i < 8; ++i) {
      a[s][i]         = (_Float16)(ag[s * 32 + aoff + i] * inv);
      a[s][i + 8]     = (_Float16)(ag[s * 32 + aoff + 8 + i] * inv);
      a[s + 2][i]     = (_Float16)hr[s * 32 + aoff + i];
      a[s + 2][i + 8] = (_Float16)hr[s * 32 + aoff + 8 + i];
    }
  }

  v8f acc[2] = {};
#pragma unroll
  for (int nt = 0; nt < 2; ++nt) {
    const _Float16* wr = wc2 + (nt * 16 + row) * 128;
#pragma unroll
    for (int s = 0; s < 4; ++s) {
      v16h b = *(const v16h*)(wr + s * 32 + kh);
      acc[nt] = WMMA(a[s], b, acc[nt]);
    }
  }

  const int mo = hi ? 8 : 0;
#pragma unroll
  for (int nt = 0; nt < 2; ++nt) {
    const int col = nt * 16 + row;
    const float bv = bias2[col];
#pragma unroll
    for (int j = 0; j < 8; ++j) {
      const int m = base + mo + j;
      h2[m * 32 + col] = fmaxf(acc[nt][j] + bv, 0.0f);
    }
  }
}

__global__ void k_out(const float* __restrict__ h2, const float* __restrict__ Wout,
                      const float* __restrict__ bout, float* __restrict__ out, int N) {
  int i = blockIdx.x * blockDim.x + threadIdx.x;
  if (i >= N) return;
  const float* hr = h2 + (size_t)i * 32;
  float acc = bout[0];
#pragma unroll
  for (int c = 0; c < 32; ++c) acc += hr[c] * Wout[c];
  out[i] = acc;
}

extern "C" void kernel_launch(void* const* d_in, const int* in_sizes, int n_in,
                              void* d_out, int out_size, void* d_ws, size_t ws_size,
                              hipStream_t stream) {
  const float* x    = (const float*)d_in[0];
  const int*   ei   = (const int*)d_in[1];
  const float* W1l  = (const float*)d_in[2];
  const float* W1r  = (const float*)d_in[3];
  const float* b1   = (const float*)d_in[4];
  const float* W2l  = (const float*)d_in[5];
  const float* W2r  = (const float*)d_in[6];
  const float* b2   = (const float*)d_in[7];
  const float* Wout = (const float*)d_in[8];
  const float* bout = (const float*)d_in[9];

  const int N = in_sizes[0] / 30;   // 100000 (divisible by 16)
  const int E = in_sizes[1] / 2;    // 3200000
  const int* src = ei;
  const int* dst = ei + E;

  // Workspace layout (float units, all regions 128B-aligned): cnt | agg1(Nx32) | h1(Nx64)
  // | agg2(Nx64) | wc1(f16 64x64) | wc2(f16 32x128).  h2 reuses agg1 after layer 1.
  float* ws = (float*)d_ws;
  const size_t nA      = (((size_t)N + 1023) / 1024) * 1024;
  const size_t offAgg1 = nA;
  const size_t offH1   = offAgg1 + (size_t)N * 32;
  const size_t offAgg2 = offH1 + (size_t)N * 64;
  const size_t offW1   = offAgg2 + (size_t)N * 64;
  const size_t offW2   = offW1 + 2048;
  const size_t needF   = offW2 + 2048;        // ~16.1M floats ~= 64.4 MB
  if (ws_size < needF * sizeof(float)) return;

  float* cnt  = ws;
  float* agg1 = ws + offAgg1;
  float* h1   = ws + offH1;
  float* agg2 = ws + offAgg2;
  float* h2   = ws + offAgg1;                 // reuse (agg1 dead after k_gemm1)
  _Float16* wc1 = (_Float16*)(ws + offW1);
  _Float16* wc2 = (_Float16*)(ws + offW2);

  const int z1 = (int)offH1;                  // cnt(+pad) + agg1
  const int z2 = N * 64;                      // agg2
  k_zero<<<(z1 + 255) / 256, 256, 0, stream>>>(ws, z1);
  k_zero<<<(z2 + 255) / 256, 256, 0, stream>>>(agg2, z2);
  k_prep<<<1, 256, 0, stream>>>(W1l, W1r, W2l, W2r, wc1, wc2);

  const int t1 = E * 32;                      // 102.4M lanes
  k_scatter1<<<(t1 + 255) / 256, 256, 0, stream>>>(x, src, dst, agg1, cnt, t1);

  const int ntiles = N / 16;
  const int gblocks = (ntiles + 7) / 8;       // 8 waves (tiles) per 256-thread block
  k_gemm1<<<gblocks, 256, 0, stream>>>(x, agg1, cnt, wc1, b1, h1, ntiles);

  const int t2 = E * 64;                      // 204.8M lanes
  k_scatter2<<<(t2 + 255) / 256, 256, 0, stream>>>(h1, src, dst, agg2, t2);

  k_gemm2<<<gblocks, 256, 0, stream>>>(h1, agg2, cnt, wc2, b2, h2, ntiles);

  k_out<<<(N + 255) / 256, 256, 0, stream>>>(h2, Wout, bout, (float*)d_out, N);
}
